// Graph_32298154066287
// MI455X (gfx1250) — matrix operand order repaired and checked by reading, not compiled, not run on hardware
//
#include <hip/hip_runtime.h>

// GCN forward on MI455X (gfx1250, wave32, bf16 WMMA + TDM async B-panel staging).
// adj (256 MB f32) is streamed twice -> ~22us HBM floor @ 23.3 TB/s.
// bf16 WMMA w/ f32 accumulation keeps matrix math below that floor.
// The shared B panel (supportT / hW2T) is DMA'd into LDS once per block via
// TENSOR_LOAD_TO_LDS (double buffered, TENSORcnt), so the vector-memory path
// is dedicated to the adj stream and B comes from LDS (ds_load_b128).

typedef __bf16 bf16_t;
typedef __attribute__((ext_vector_type(16))) __bf16 v16bf;
typedef __attribute__((ext_vector_type(8)))  __bf16 v8bf;
typedef __attribute__((ext_vector_type(8)))  float  v8f;
typedef __attribute__((ext_vector_type(4)))  float  v4f;
typedef __attribute__((ext_vector_type(4)))  unsigned int u32x4;
typedef __attribute__((ext_vector_type(4)))  int    i32x4;
typedef __attribute__((ext_vector_type(8)))  int    i32x8;

#define NROW   8192
#define NFEAT  512
#define NHID   128
#define NCLS   40
#define NCLSP  48      // classes padded to 3 x 16 WMMA tiles
#define KSPLIT 4       // K-split factor for the two adj GEMMs
#define KSLAB  64      // K extent per TDM slab (2 WMMA K-steps)
// LDS panel: per column KSLAB*2 = 128 B data + 32 B TDM pad -> 160 B stride
// (16B-aligned ds_load_b128, <=2-way bank conflicts)
#define PANEL_STRIDE 160

// ---------------- fragment loaders (layouts per CDNA5 ISA 7.12.2) ----------

// A 16x32 bf16 frag from row-major f32 source (convert in registers).
// lane 0-15: row=lane, K = {0..7, 16..23}; lane 16-31: row=lane-16, K={8..15,24..31}
__device__ __forceinline__ v16bf load_a_from_f32(const float* __restrict__ A,
                                                 int lda, int row, int k0, int half) {
  const float* p = A + (size_t)row * lda + (k0 + half * 8);
  v4f f0 = *(const v4f*)(p);
  v4f f1 = *(const v4f*)(p + 4);
  v4f f2 = *(const v4f*)(p + 16);
  v4f f3 = *(const v4f*)(p + 20);
  v16bf a;
#pragma unroll
  for (int i = 0; i < 4; ++i) {
    a[i]      = (__bf16)f0[i];
    a[4 + i]  = (__bf16)f1[i];
    a[8 + i]  = (__bf16)f2[i];
    a[12 + i] = (__bf16)f3[i];
  }
  return a;
}

// A 16x32 bf16 frag from row-major bf16 source.
__device__ __forceinline__ v16bf load_a_from_bf16(const bf16_t* __restrict__ A,
                                                  int lda, int row, int k0, int half) {
  const bf16_t* p = A + (size_t)row * lda + (k0 + half * 8);
  v8bf g0 = *(const v8bf*)(p);
  v8bf g1 = *(const v8bf*)(p + 16);
  v16bf a;
#pragma unroll
  for (int i = 0; i < 8; ++i) { a[i] = g0[i]; a[8 + i] = g1[i]; }
  return a;
}

// B 32x16 bf16 frag from TRANSPOSED global source BT[col][k] (ld = ldk).
// lane 0-15: col=lane, K = {0..7, 16..23}; lane 16-31: col=lane-16, K={8..15,24..31}
__device__ __forceinline__ v16bf load_b_from_T(const bf16_t* __restrict__ BT,
                                               int ldk, int col, int k0, int half) {
  const bf16_t* p = BT + (size_t)col * ldk + (k0 + half * 8);
  v8bf g0 = *(const v8bf*)(p);
  v8bf g1 = *(const v8bf*)(p + 16);
  v16bf b;
#pragma unroll
  for (int i = 0; i < 8; ++i) { b[i] = g0[i]; b[8 + i] = g1[i]; }
  return b;
}

// B frag from the LDS panel (TDM layout: col-major, 160 B col stride).
__device__ __forceinline__ v16bf load_b_from_lds(const char* buf, int col,
                                                 int kloc, int half) {
  const char* p = buf + col * PANEL_STRIDE + (kloc + half * 8) * 2;
  v8bf g0 = *(const v8bf*)(p);
  v8bf g1 = *(const v8bf*)(p + 32);
  v16bf b;
#pragma unroll
  for (int i = 0; i < 8; ++i) { b[i] = g0[i]; b[8 + i] = g1[i]; }
  return b;
}

// ---------------- TDM slab issue (CDNA5 ISA ch.8 descriptor) ---------------
// Loads a KSLAB x ncols slab of BT (bf16, ld=ldk elems) into LDS at ldsAddr,
// padding 32 B after each 128 B row => PANEL_STRIDE per column.
__device__ __forceinline__ void tdm_issue_slab(const bf16_t* slabPtr,
                                               unsigned int ldsAddr,
                                               int ldk, int ncols) {
  unsigned long long ga = (unsigned long long)(uintptr_t)slabPtr;
  u32x4 g0 = { 1u,                              // count=1 user descriptor
               ldsAddr,                         // LDS byte address
               (unsigned int)ga,                // global_addr[31:0]
               (unsigned int)((ga >> 32) & 0x01FFFFFFu) | (2u << 30) }; // type=2
  i32x8 g1 = { (1 << 16)      // data_size = 2 bytes
             | (1 << 20)      // pad_enable
             | (4 << 22)      // pad_interval: 32 DWORDs (=128 B = one tile row)
             | (7 << 25),     // pad_amount: 8 DWORDs (=32 B)
               (int)(((unsigned)ldk & 0xFFFFu) << 16),              // td0 lo16
               (int)(((unsigned)ldk >> 16) | (((unsigned)ncols & 0xFFFFu) << 16)),
               (int)(((unsigned)ncols >> 16) | ((unsigned)KSLAB << 16)), // tile_dim0
               ncols,                                               // tile_dim1
               ldk,                                                 // stride0 lo32
               0, 0 };
  i32x4 gz = { 0, 0, 0, 0 };
#if defined(__clang_major__) && (__clang_major__ >= 23)
  i32x8 gz8 = { 0, 0, 0, 0, 0, 0, 0, 0 };
  __builtin_amdgcn_tensor_load_to_lds(g0, g1, gz, gz, gz8, 0);
#else
  __builtin_amdgcn_tensor_load_to_lds(g0, g1, gz, gz, 0);
#endif
}

// ---------------- GEMM kernels --------------------------------------------
// Block = 256 threads = 8 waves; wave w owns rows [ (bx*8+w)*16, +16 ).
// Each wave covers the FULL N extent (NT tiles) so adj is read exactly once.

// C = A(f32, row-major) * BT(bf16)^T, full K, store bf16 TRANSPOSED OutT[col][m].
template <int NT>
__global__ void gemm_f32A_storeT(const float* __restrict__ A,
                                 const bf16_t* __restrict__ BT,
                                 bf16_t* __restrict__ OutT, int lda, int K) {
  const int lane = threadIdx.x & 31;
  const int wave = threadIdx.x >> 5;
  const int lr = lane & 15, half = lane >> 4;
  const int mBase = (blockIdx.x * 8 + wave) * 16;
  const int row = mBase + lr;
  v8f acc[NT] = {};
  for (int k0 = 0; k0 < K; k0 += 32) {
    v16bf a = load_a_from_f32(A, lda, row, k0, half);
#pragma unroll
    for (int t = 0; t < NT; ++t) {
      v16bf b = load_b_from_T(BT, K, t * 16 + lr, k0, half);
      acc[t] = __builtin_amdgcn_wmma_f32_16x16x32_bf16(false, a, false, b,
                                                       (short)0, acc[t], false, false);
    }
  }
#pragma unroll
  for (int t = 0; t < NT; ++t) {
    v8bf o;
#pragma unroll
    for (int r = 0; r < 8; ++r) o[r] = (__bf16)acc[t][r];
    *(v8bf*)(OutT + (size_t)(t * 16 + lr) * NROW + mBase + half * 8) = o;
  }
}

// Same but A is bf16 row-major (used for h @ W2).
template <int NT>
__global__ void gemm_bf16A_storeT(const bf16_t* __restrict__ A,
                                  const bf16_t* __restrict__ BT,
                                  bf16_t* __restrict__ OutT, int lda, int K) {
  const int lane = threadIdx.x & 31;
  const int wave = threadIdx.x >> 5;
  const int lr = lane & 15, half = lane >> 4;
  const int mBase = (blockIdx.x * 8 + wave) * 16;
  const int row = mBase + lr;
  v8f acc[NT] = {};
  for (int k0 = 0; k0 < K; k0 += 32) {
    v16bf a = load_a_from_bf16(A, lda, row, k0, half);
#pragma unroll
    for (int t = 0; t < NT; ++t) {
      v16bf b = load_b_from_T(BT, K, t * 16 + lr, k0, half);
      acc[t] = __builtin_amdgcn_wmma_f32_16x16x32_bf16(false, a, false, b,
                                                       (short)0, acc[t], false, false);
    }
  }
#pragma unroll
  for (int t = 0; t < NT; ++t) {
    v8bf o;
#pragma unroll
    for (int r = 0; r < 8; ++r) o[r] = (__bf16)acc[t][r];
    *(v8bf*)(OutT + (size_t)(t * 16 + lr) * NROW + mBase + half * 8) = o;
  }
}

// K-split adj GEMM with TDM-staged, double-buffered B panel in LDS.
// Cbuf(f32, row-major, ld=ldc) += A_chunk * BT_chunk via hardware f32 atomics.
template <int NT>
__global__ void gemm_adj_tdm(const float* __restrict__ A,
                             const bf16_t* __restrict__ BT,
                             float* __restrict__ Cbuf,
                             int lda, int ldk, int K, int ldc) {
  extern __shared__ char smem[];
  const int lane = threadIdx.x & 31;
  const int wave = threadIdx.x >> 5;
  const int lr = lane & 15, half = lane >> 4;
  const int mBase = (blockIdx.x * 8 + wave) * 16;
  const int row = mBase + lr;
  const int kChunk = K / gridDim.y;
  const int kStart = blockIdx.y * kChunk;
  const int nSlab = kChunk / KSLAB;
  const unsigned int panelBytes = NT * 16 * PANEL_STRIDE;
  // dynamic-LDS base offset (wave-relative byte address for the TDM D#)
  const unsigned int ldsBase = __builtin_amdgcn_groupstaticsize();

  v8f acc[NT] = {};

  if (wave == 0)
    tdm_issue_slab(BT + (size_t)kStart, ldsBase, ldk, NT * 16);

  for (int s = 0; s < nSlab; ++s) {
    if (wave == 0 && (s + 1) < nSlab)
      tdm_issue_slab(BT + (size_t)(kStart + (s + 1) * KSLAB),
                     ldsBase + (unsigned)((s + 1) & 1) * panelBytes, ldk, NT * 16);
    // wave0: wait for the older in-flight slab; no-op for other waves
    if ((s + 1) < nSlab) __builtin_amdgcn_s_wait_tensorcnt(1);
    else                 __builtin_amdgcn_s_wait_tensorcnt(0);
    __syncthreads();                     // publish slab s to all waves
    const char* buf = smem + (size_t)(s & 1) * panelBytes;
    const int kb = kStart + s * KSLAB;
#pragma unroll
    for (int kk = 0; kk < KSLAB; kk += 32) {
      v16bf a = load_a_from_f32(A, lda, row, kb + kk, half);
#pragma unroll
      for (int t = 0; t < NT; ++t) {
        v16bf b = load_b_from_lds(buf, t * 16 + lr, kk, half);
        acc[t] = __builtin_amdgcn_wmma_f32_16x16x32_bf16(false, a, false, b,
                                                         (short)0, acc[t], false, false);
      }
    }
    __syncthreads();                     // all reads of buf done before reuse
  }

#pragma unroll
  for (int t = 0; t < NT; ++t) {
    const int col = t * 16 + lr;
#pragma unroll
    for (int r = 0; r < 8; ++r) {
      unsafeAtomicAdd(&Cbuf[(size_t)(mBase + half * 8 + r) * ldc + col], acc[t][r]);
    }
  }
}

// ---------------- small helper kernels ------------------------------------

__global__ void zero_kernel(float* __restrict__ p, int n) {
  int i = blockIdx.x * blockDim.x + threadIdx.x;
  int stride = gridDim.x * blockDim.x;
  for (; i < n; i += stride) p[i] = 0.f;
}

// W1 [NFEAT][NHID] f32 -> W1T [NHID][NFEAT] bf16
__global__ void conv_w1t(const float* __restrict__ W1, bf16_t* __restrict__ W1T) {
  int idx = blockIdx.x * blockDim.x + threadIdx.x;
  if (idx >= NFEAT * NHID) return;
  int f = idx / NHID, h = idx % NHID;
  W1T[(size_t)h * NFEAT + f] = (__bf16)W1[idx];
}

// W2 [NHID][NCLS] f32 -> W2T [NCLSP][NHID] bf16, zero-padded cols 40..47
__global__ void conv_w2t(const float* __restrict__ W2, bf16_t* __restrict__ W2T) {
  int idx = blockIdx.x * blockDim.x + threadIdx.x;
  if (idx >= NCLSP * NHID) return;
  int c = idx / NHID, k = idx % NHID;
  W2T[idx] = (c < NCLS) ? (__bf16)W2[(size_t)k * NCLS + c] : (__bf16)0.f;
}

// h = relu(hbuf + b1), f32 -> bf16 row-major [NROW][NHID]
__global__ void epilogue_h_kernel(const float* __restrict__ hbuf,
                                  const float* __restrict__ b1,
                                  bf16_t* __restrict__ h) {
  int idx = blockIdx.x * blockDim.x + threadIdx.x;
  if (idx >= NROW * NHID) return;
  int c = idx & (NHID - 1);
  float v = hbuf[idx] + b1[c];
  h[idx] = (__bf16)fmaxf(v, 0.f);
}

// out = log_softmax(lbuf[:, :40] + b2) per row
__global__ void logsoftmax_kernel(const float* __restrict__ lbuf,
                                  const float* __restrict__ b2,
                                  float* __restrict__ out) {
  int row = blockIdx.x * blockDim.x + threadIdx.x;
  if (row >= NROW) return;
  float v[NCLS];
  float mx = -3.0e38f;
  for (int c = 0; c < NCLS; ++c) {
    v[c] = lbuf[(size_t)row * NCLSP + c] + b2[c];
    mx = fmaxf(mx, v[c]);
  }
  float s = 0.f;
  for (int c = 0; c < NCLS; ++c) s += expf(v[c] - mx);
  float lse = mx + logf(s);
  for (int c = 0; c < NCLS; ++c) out[(size_t)row * NCLS + c] = v[c] - lse;
}

// ---------------- launch ---------------------------------------------------

extern "C" void kernel_launch(void* const* d_in, const int* in_sizes, int n_in,
                              void* d_out, int out_size, void* d_ws, size_t ws_size,
                              hipStream_t stream) {
  const float* x   = (const float*)d_in[0];   // [8192][512]
  const float* adj = (const float*)d_in[1];   // [8192][8192]
  const float* W1  = (const float*)d_in[2];   // [512][128]
  const float* b1  = (const float*)d_in[3];   // [128]
  const float* W2  = (const float*)d_in[4];   // [128][40]
  const float* b2  = (const float*)d_in[5];   // [40]
  float* out = (float*)d_out;                 // [8192][40] log_softmax

  // workspace carve-out (~10.5 MB, all 16B-aligned)
  float*  hbuf     = (float*)d_ws;                      // NROW*NHID   f32
  float*  lbuf     = hbuf + (size_t)NROW * NHID;        // NROW*NCLSP  f32
  bf16_t* supportT = (bf16_t*)(lbuf + (size_t)NROW * NCLSP); // [NHID][NROW]
  bf16_t* hmat     = supportT + (size_t)NHID * NROW;    // [NROW][NHID]
  bf16_t* hw2T     = hmat + (size_t)NROW * NHID;        // [NCLSP][NROW]
  bf16_t* W1T      = hw2T + (size_t)NCLSP * NROW;       // [NHID][NFEAT]
  bf16_t* W2T      = W1T + (size_t)NHID * NFEAT;        // [NCLSP][NHID]

  // 0) zero the atomic accumulation buffers (hbuf, lbuf are contiguous)
  zero_kernel<<<512, 256, 0, stream>>>(hbuf, NROW * NHID + NROW * NCLSP);

  // 1) weight transpose+convert to bf16
  conv_w1t<<<(NFEAT * NHID + 255) / 256, 256, 0, stream>>>(W1, W1T);
  conv_w2t<<<(NCLSP * NHID + 255) / 256, 256, 0, stream>>>(W2, W2T);

  // 2) supportT = (x @ W1)^T   (M=8192, K=512, N=128)
  gemm_f32A_storeT<NHID / 16><<<NROW / 128, 256, 0, stream>>>(x, W1T, supportT,
                                                              NFEAT, NFEAT);

  // 3) hbuf += adj @ support   (M=8192, K=8192, N=128, K-split x4, TDM B panel)
  gemm_adj_tdm<NHID / 16>
      <<<dim3(NROW / 128, KSPLIT), 256, 2u * (NHID / 16) * 16 * PANEL_STRIDE, stream>>>(
          adj, supportT, hbuf, NROW, NROW, NROW, NHID);

  // 4) h = relu(hbuf + b1) -> bf16
  epilogue_h_kernel<<<(NROW * NHID + 255) / 256, 256, 0, stream>>>(hbuf, b1, hmat);

  // 5) hw2T = (h @ W2)^T       (M=8192, K=128, N=48 padded)
  gemm_bf16A_storeT<NCLSP / 16><<<NROW / 128, 256, 0, stream>>>(hmat, W2T, hw2T,
                                                                NHID, NHID);

  // 6) lbuf += adj @ hw2       (M=8192, K=8192, N=48, K-split x4, TDM B panel)
  gemm_adj_tdm<NCLSP / 16>
      <<<dim3(NROW / 128, KSPLIT), 256, 2u * (NCLSP / 16) * 16 * PANEL_STRIDE, stream>>>(
          adj, hw2T, lbuf, NROW, NROW, NROW, NCLSP);

  // 7) out = log_softmax(lbuf + b2)
  logsoftmax_kernel<<<NROW / 256, 256, 0, stream>>>(lbuf, b2, out);
}
